// InternVisionAttention_31688268710153
// MI455X (gfx1250) — matrix operand order repaired and checked by reading, not compiled, not run on hardware
//
#include <hip/hip_runtime.h>
#include <hip/hip_bf16.h>
#include <stdint.h>

// ---------------------------------------------------------------------------
// CDNA5 (gfx1250) InternVision attention: QKV GEMM -> RMSNorm -> RoPE ->
// block-diagonal flash attention -> proj GEMM, all matmuls on
// v_wmma_f32_16x16x32_f16; GEMM tiles staged with global_load_async_to_lds_b128
// (ASYNCcnt) into double-buffered LDS.
// Problem: S=2048, E=1024, H=16, D=64, segments of 512.
// ---------------------------------------------------------------------------

typedef __attribute__((ext_vector_type(16))) _Float16 v16h;
typedef __attribute__((ext_vector_type(8)))  float    v8f;

union Frag16 {
  v16h h;
  uint4 u[2];
};

__device__ __forceinline__ v8f wmma_f16f32(v16h a, v16h b, v8f c) {
  // D = A(16x32 f16) * B(32x16 f16) + C(16x16 f32)
  return __builtin_amdgcn_wmma_f32_16x16x32_f16(
      /*neg_a=*/false, a, /*neg_b=*/false, b,
      /*c_mod=*/(short)0, c, /*reuse_a=*/false, /*reuse_b=*/false);
}

// Async DMA: 16 bytes global -> LDS, tracked by ASYNCcnt (no VGPR staging).
__device__ __forceinline__ void async_b128_to_lds(const void* gaddr, void* ldsaddr) {
  asm volatile("global_load_async_to_lds_b128 %0, %1, off"
               :
               : "v"((uint32_t)(uintptr_t)ldsaddr), "v"(gaddr)
               : "memory");
}

__device__ __forceinline__ void wait_async0() {
  asm volatile("s_wait_asynccnt 0x0" ::: "memory");
}

#define kS 2048
#define kE 1024
#define kH 16
#define kD 64

// ---------------------------------------------------------------------------
// f32 -> f16 cast (weights / activations), n must be a multiple of 4
// ---------------------------------------------------------------------------
__global__ __launch_bounds__(256) void cast_f32_to_f16_kernel(
    const float* __restrict__ src, _Float16* __restrict__ dst, int n) {
  int idx = (blockIdx.x * blockDim.x + threadIdx.x) * 4;
  if (idx < n) {
    float4 v = *(const float4*)(src + idx);
    dst[idx + 0] = (_Float16)v.x;
    dst[idx + 1] = (_Float16)v.y;
    dst[idx + 2] = (_Float16)v.z;
    dst[idx + 3] = (_Float16)v.w;
  }
}

// ---------------------------------------------------------------------------
// WMMA GEMM: C[M][N] = A[M][K] * B[N][K]^T + bias[N]
// block tile 128x128, BK=32, 8 waves (2x4), each wave: 64x32 = 4x2 WMMA tiles.
// A/B tiles DMA'd into double-buffered LDS with async-to-LDS; tile i+1's DMA
// overlaps tile i's WMMAs. M,N multiples of 128; K multiple of 32.
// ---------------------------------------------------------------------------
#define BM 128
#define BN 128
#define BK 32

__global__ __launch_bounds__(256) void wmma_gemm_bias_kernel(
    const _Float16* __restrict__ A,    // [M][K]
    const _Float16* __restrict__ B,    // [N][K]   (weight rows = output cols)
    const float*    __restrict__ bias, // [N]
    float*          __restrict__ C,    // [M][N]
    int M, int N, int K) {
  __shared__ _Float16 As[2][BM][BK];
  __shared__ _Float16 Bs[2][BM][BK];

  const int tid  = threadIdx.x;
  const int lane = tid & 31;
  const int wave = tid >> 5;
  const int wm   = wave >> 2;  // 0..1  (64-row strip)
  const int wn   = wave & 3;   // 0..3  (32-col strip)
  const int hs   = lane >> 4;  // half-wave select
  const int lr   = lane & 15;

  const int tileM = blockIdx.x * BM;
  const int tileN = blockIdx.y * BN;

  // per-thread staging coordinates: 2 x b128 per matrix per tile
  const int u0 = tid * 2;
  const int r0 = u0 >> 2, c0 = (u0 & 3) * 8;
  const int u1 = u0 + 1;
  const int r1 = u1 >> 2, c1 = (u1 & 3) * 8;

  const _Float16* Arow0 = A + (size_t)(tileM + r0) * K + c0;
  const _Float16* Arow1 = A + (size_t)(tileM + r1) * K + c1;
  const _Float16* Brow0 = B + (size_t)(tileN + r0) * K + c0;
  const _Float16* Brow1 = B + (size_t)(tileN + r1) * K + c1;

  v8f acc[4][2];
#pragma unroll
  for (int mi = 0; mi < 4; ++mi)
#pragma unroll
    for (int nj = 0; nj < 2; ++nj)
#pragma unroll
      for (int r = 0; r < 8; ++r) acc[mi][nj][r] = 0.0f;

  // prologue: DMA tile 0 into buffer 0
  async_b128_to_lds(Arow0, &As[0][r0][c0]);
  async_b128_to_lds(Brow0, &Bs[0][r0][c0]);
  async_b128_to_lds(Arow1, &As[0][r1][c1]);
  async_b128_to_lds(Brow1, &Bs[0][r1][c1]);

  int buf = 0;
  for (int k0 = 0; k0 < K; k0 += BK) {
    // current buffer's DMA complete (mine), then everyone's (barrier); the
    // barrier also guarantees all waves are done reading buf^1.
    wait_async0();
    __syncthreads();

    if (k0 + BK < K) {  // kick off next tile's DMA into the other buffer
      const int kn = k0 + BK;
      const int nb = buf ^ 1;
      async_b128_to_lds(Arow0 + kn, &As[nb][r0][c0]);
      async_b128_to_lds(Brow0 + kn, &Bs[nb][r0][c0]);
      async_b128_to_lds(Arow1 + kn, &As[nb][r1][c1]);
      async_b128_to_lds(Brow1 + kn, &Bs[nb][r1][c1]);
    }
    if (k0 + 2 * BK < K) {  // warm L2 two tiles ahead (global_prefetch_b8)
      __builtin_prefetch(Arow0 + k0 + 2 * BK, 0, 1);
      __builtin_prefetch(Brow0 + k0 + 2 * BK, 0, 1);
    }

    // A fragment (16x32): lanes 0-15 hold K {0..7,16..23}; lanes 16-31 {8..15,24..31}
    Frag16 af[4], bf[2];
#pragma unroll
    for (int mi = 0; mi < 4; ++mi) {
      int row = wm * 64 + mi * 16 + lr;
      int kb  = hs * 8;
      af[mi].u[0] = *(const uint4*)&As[buf][row][kb];
      af[mi].u[1] = *(const uint4*)&As[buf][row][kb + 16];
    }
    // B fragment (32x16): lanes 0-15 hold K {0..7,16..23}; lanes 16-31 {8..15,24..31}
#pragma unroll
    for (int nj = 0; nj < 2; ++nj) {
      int row = wn * 32 + nj * 16 + lr;
      int kb  = hs * 8;
      bf[nj].u[0] = *(const uint4*)&Bs[buf][row][kb];
      bf[nj].u[1] = *(const uint4*)&Bs[buf][row][kb + 16];
    }
#pragma unroll
    for (int mi = 0; mi < 4; ++mi)
#pragma unroll
      for (int nj = 0; nj < 2; ++nj)
        acc[mi][nj] = wmma_f16f32(af[mi].h, bf[nj].h, acc[mi][nj]);

    buf ^= 1;
  }

  // C layout: VGPR r -> row (r + 8*hs), col = lr within each 16x16 tile.
#pragma unroll
  for (int mi = 0; mi < 4; ++mi)
#pragma unroll
    for (int nj = 0; nj < 2; ++nj) {
      int col  = tileN + wn * 32 + nj * 16 + lr;
      float bv = bias[col];
#pragma unroll
      for (int r = 0; r < 8; ++r) {
        int row = tileM + wm * 64 + mi * 16 + r + 8 * hs;
        C[(size_t)row * N + col] = acc[mi][nj][r] + bv;
      }
    }
}

// ---------------------------------------------------------------------------
// RMSNorm (over full E per token, matching reference) + RoPE, emits:
//   qf16[h][s][d], kf16[h][s][d]  (row-major in d -> cheap Q/K fragments)
//   vt16[h][d][s]                 (keys contiguous -> cheap PV B fragments)
// One 256-thread block per token.
// ---------------------------------------------------------------------------
__global__ __launch_bounds__(256) void norm_rope_kernel(
    const float* __restrict__ qkv,  // [S][3*E]  (q | k | v)
    const float* __restrict__ rot,  // [S][32]
    const float* __restrict__ qw,   // [E]
    const float* __restrict__ kw,   // [E]
    _Float16* __restrict__ qf, _Float16* __restrict__ kf,
    _Float16* __restrict__ vt) {
  const int s    = blockIdx.x;
  const int tid  = threadIdx.x;
  const int lane = tid & 31;
  const int wave = tid >> 5;

  __shared__ float qn[kE];
  __shared__ float kn[kE];
  __shared__ float red[16];

  const float* row = qkv + (size_t)s * (3 * kE);
  const int e0 = tid * 4;
  float4 qv = *(const float4*)(row + e0);
  float4 kv = *(const float4*)(row + kE + e0);

  float sq = qv.x * qv.x + qv.y * qv.y + qv.z * qv.z + qv.w * qv.w;
  float sk = kv.x * kv.x + kv.y * kv.y + kv.z * kv.z + kv.w * kv.w;
#pragma unroll
  for (int m = 16; m >= 1; m >>= 1) {
    sq += __shfl_xor(sq, m, 32);
    sk += __shfl_xor(sk, m, 32);
  }
  if (lane == 0) { red[wave] = sq; red[8 + wave] = sk; }
  __syncthreads();
  float tq = 0.f, tk = 0.f;
#pragma unroll
  for (int i = 0; i < 8; ++i) { tq += red[i]; tk += red[8 + i]; }
  const float rq = rsqrtf(tq * (1.0f / kE) + 1e-6f);
  const float rk = rsqrtf(tk * (1.0f / kE) + 1e-6f);

  qn[e0 + 0] = qv.x * rq * qw[e0 + 0];
  qn[e0 + 1] = qv.y * rq * qw[e0 + 1];
  qn[e0 + 2] = qv.z * rq * qw[e0 + 2];
  qn[e0 + 3] = qv.w * rq * qw[e0 + 3];
  kn[e0 + 0] = kv.x * rk * kw[e0 + 0];
  kn[e0 + 1] = kv.y * rk * kw[e0 + 1];
  kn[e0 + 2] = kv.z * rk * kw[e0 + 2];
  kn[e0 + 3] = kv.w * rk * kw[e0 + 3];
  __syncthreads();

#pragma unroll
  for (int j = 0; j < 4; ++j) {
    int e = e0 + j;
    int h = e >> 6;
    int d = e & 63;
    float fr = rot[(size_t)s * 32 + (d & 31)];
    float c  = __cosf(fr);
    float sn = __sinf(fr);
    float rhq = (d < 32) ? -qn[e + 32] : qn[e - 32];
    float rhk = (d < 32) ? -kn[e + 32] : kn[e - 32];
    size_t oidx = ((size_t)h * kS + s) * kD + d;
    qf[oidx] = (_Float16)(qn[e] * c + rhq * sn);
    kf[oidx] = (_Float16)(kn[e] * c + rhk * sn);
    vt[((size_t)h * kD + d) * kS + s] = (_Float16)row[2 * kE + e];
  }
}

// ---------------------------------------------------------------------------
// Flash attention, block-diagonal segments. One wave (32 threads) per
// (head, 16-query tile). Per 32-key chunk: 4 score WMMAs (K=d=64 -> 2 steps,
// 2 key tiles), online softmax (f32), P -> LDS -> f16 A-fragment, 4 PV WMMAs.
// ---------------------------------------------------------------------------
__global__ __launch_bounds__(32) void flash_attn_kernel(
    const _Float16* __restrict__ qf,   // [H][S][D]
    const _Float16* __restrict__ kfm,  // [H][S][D]
    const _Float16* __restrict__ vt,   // [H][D][S]
    const int* __restrict__ cu,        // [5]
    _Float16* __restrict__ aout) {     // [S][E]
  const int h    = blockIdx.y;
  const int q0   = blockIdx.x * 16;
  const int lane = threadIdx.x;
  const int hs   = lane >> 4;
  const int lr   = lane & 15;

  __shared__ _Float16 pbuf[16][32];

  int seg_start = 0, seg_end = kS;
#pragma unroll
  for (int i = 0; i < 4; ++i)
    if (q0 >= cu[i] && q0 < cu[i + 1]) { seg_start = cu[i]; seg_end = cu[i + 1]; }

  // Q fragments for d=0..31 and d=32..63 (held in registers for whole sweep)
  Frag16 aq0, aq1;
  {
    const _Float16* rp = qf + ((size_t)h * kS + q0 + lr) * kD;
    aq0.u[0] = *(const uint4*)(rp + hs * 8);
    aq0.u[1] = *(const uint4*)(rp + 16 + hs * 8);
    aq1.u[0] = *(const uint4*)(rp + 32 + hs * 8);
    aq1.u[1] = *(const uint4*)(rp + 48 + hs * 8);
  }

  v8f o[4];
  float mrow[8], lrow[8];
#pragma unroll
  for (int t = 0; t < 4; ++t)
#pragma unroll
    for (int r = 0; r < 8; ++r) o[t][r] = 0.0f;
#pragma unroll
  for (int r = 0; r < 8; ++r) { mrow[r] = -3.0e38f; lrow[r] = 0.0f; }

  const float scale = 0.125f;  // 1/sqrt(64)

  for (int kb = seg_start; kb < seg_end; kb += 32) {
    // ---- scores: two 16-key tiles, K-dim = d (two 32-wide steps) ----
    v8f sf[2];
#pragma unroll
    for (int j = 0; j < 2; ++j) {
      const _Float16* kp = kfm + ((size_t)h * kS + kb + j * 16 + lr) * kD;
      Frag16 b0, b1;
      b0.u[0] = *(const uint4*)(kp + hs * 8);
      b0.u[1] = *(const uint4*)(kp + hs * 8 + 16);
      b1.u[0] = *(const uint4*)(kp + 32 + hs * 8);
      b1.u[1] = *(const uint4*)(kp + 32 + hs * 8 + 16);
      v8f t;
#pragma unroll
      for (int r = 0; r < 8; ++r) t[r] = 0.0f;
      t = wmma_f16f32(aq0.h, b0.h, t);
      t = wmma_f16f32(aq1.h, b1.h, t);
      sf[j] = t;
    }
    // scale + tail mask (per-lane uniform: this lane's score column)
#pragma unroll
    for (int j = 0; j < 2; ++j) {
      bool oob = (kb + j * 16 + lr) >= seg_end;
#pragma unroll
      for (int r = 0; r < 8; ++r) {
        float v = sf[j][r] * scale;
        sf[j][r] = oob ? -3.0e38f : v;
      }
    }
    // ---- online softmax: row stats reduced across each 16-lane half ----
    float cm[8];
#pragma unroll
    for (int r = 0; r < 8; ++r) cm[r] = fmaxf(sf[0][r], sf[1][r]);
#pragma unroll
    for (int m = 8; m >= 1; m >>= 1)
#pragma unroll
      for (int r = 0; r < 8; ++r) cm[r] = fmaxf(cm[r], __shfl_xor(cm[r], m, 32));

    float pr0[8], pr1[8], rs[8];
#pragma unroll
    for (int r = 0; r < 8; ++r) {
      float mn   = fmaxf(mrow[r], cm[r]);
      float corr = __expf(mrow[r] - mn);
      mrow[r] = mn;
      lrow[r] *= corr;
#pragma unroll
      for (int t = 0; t < 4; ++t) o[t][r] *= corr;
      pr0[r] = __expf(sf[0][r] - mn);
      pr1[r] = __expf(sf[1][r] - mn);
      rs[r]  = pr0[r] + pr1[r];
    }
#pragma unroll
    for (int m = 8; m >= 1; m >>= 1)
#pragma unroll
      for (int r = 0; r < 8; ++r) rs[r] += __shfl_xor(rs[r], m, 32);
#pragma unroll
    for (int r = 0; r < 8; ++r) lrow[r] += rs[r];

    // ---- re-layout P (C-frag f32) -> 16x32 f16 tile in LDS -> A-frag ----
#pragma unroll
    for (int r = 0; r < 8; ++r) {
      int prow = r + 8 * hs;
      pbuf[prow][lr]      = (_Float16)pr0[r];
      pbuf[prow][16 + lr] = (_Float16)pr1[r];
    }
    asm volatile("s_wait_dscnt 0x0" ::: "memory");
    Frag16 pf;
    pf.u[0] = *(const uint4*)&pbuf[lr][hs * 8];
    pf.u[1] = *(const uint4*)&pbuf[lr][16 + hs * 8];
    asm volatile("s_wait_dscnt 0x0" ::: "memory");

    // ---- PV: B fragments from V^T, keys contiguous along K ----
#pragma unroll
    for (int t = 0; t < 4; ++t) {
      const _Float16* vp = vt + ((size_t)h * kD + t * 16 + lr) * kS + kb;
      Frag16 vf;
      vf.u[0] = *(const uint4*)(vp + hs * 8);
      vf.u[1] = *(const uint4*)(vp + hs * 8 + 16);
      o[t] = wmma_f16f32(pf.h, vf.h, o[t]);
    }
  }

  // ---- finalize: divide by row sums, store f16 [S][E] ----
#pragma unroll
  for (int r = 0; r < 8; ++r) {
    float invl = 1.0f / lrow[r];
    int row = q0 + r + 8 * hs;
#pragma unroll
    for (int t = 0; t < 4; ++t) {
      int col = h * kD + t * 16 + lr;
      aout[(size_t)row * kE + col] = (_Float16)(o[t][r] * invl);
    }
  }
}

// ---------------------------------------------------------------------------
// Launch
// ---------------------------------------------------------------------------
extern "C" void kernel_launch(void* const* d_in, const int* in_sizes, int n_in,
                              void* d_out, int out_size, void* d_ws, size_t ws_size,
                              hipStream_t stream) {
  const float* hidden  = (const float*)d_in[0];  // [S][E]
  const float* rot     = (const float*)d_in[1];  // [S][32]
  const float* qkv_w   = (const float*)d_in[2];  // [3E][E]
  const float* qkv_b   = (const float*)d_in[3];  // [3E]
  const float* qnw     = (const float*)d_in[4];  // [E]
  const float* knw     = (const float*)d_in[5];  // [E]
  const float* proj_w  = (const float*)d_in[6];  // [E][E]
  const float* proj_b  = (const float*)d_in[7];  // [E]
  const int*   cu      = (const int*)d_in[8];    // [5]
  float*       out     = (float*)d_out;          // [S][E]

  // workspace layout (all regions 16B-aligned by construction)
  _Float16* xh     = (_Float16*)d_ws;            // S*E        f16
  _Float16* wqkvh  = xh + (size_t)kS * kE;       // 3E*E       f16
  _Float16* wprojh = wqkvh + (size_t)3 * kE * kE;// E*E        f16
  float*    qkv    = (float*)(wprojh + (size_t)kE * kE);  // S*3E f32
  _Float16* qf     = (_Float16*)(qkv + (size_t)kS * 3 * kE); // H*S*D f16
  _Float16* kf     = qf + (size_t)kH * kS * kD;
  _Float16* vt     = kf + (size_t)kH * kS * kD;
  _Float16* aout   = vt + (size_t)kH * kS * kD;  // S*E f16

  // 1) casts to f16
  cast_f32_to_f16_kernel<<<(kS * kE) / 1024, 256, 0, stream>>>(hidden, xh, kS * kE);
  cast_f32_to_f16_kernel<<<(3 * kE * kE) / 1024, 256, 0, stream>>>(qkv_w, wqkvh, 3 * kE * kE);
  cast_f32_to_f16_kernel<<<(kE * kE) / 1024, 256, 0, stream>>>(proj_w, wprojh, kE * kE);

  // 2) QKV = X * Wqkv^T + b   (2048x3072x1024)
  wmma_gemm_bias_kernel<<<dim3(kS / BM, (3 * kE) / BN), 256, 0, stream>>>(
      xh, wqkvh, qkv_b, qkv, kS, 3 * kE, kE);

  // 3) RMSNorm + RoPE, split into q/k/v^T f16
  norm_rope_kernel<<<kS, 256, 0, stream>>>(qkv, rot, qnw, knw, qf, kf, vt);

  // 4) block-diagonal flash attention
  flash_attn_kernel<<<dim3(kS / 16, kH), 32, 0, stream>>>(qf, kf, vt, cu, aout);

  // 5) out = attn * Wproj^T + b  (2048x1024x1024)
  wmma_gemm_bias_kernel<<<dim3(kS / BM, kE / BN), 256, 0, stream>>>(
      aout, wprojh, proj_b, out, kS, kE, kE);
}
